// ResidualTokenAdapter_88622355186406
// MI455X (gfx1250) — matrix operand order repaired and checked by reading, not compiled, not run on hardware
//
#include <hip/hip_runtime.h>

// ---------------------------------------------------------------------------
// Fused residual token adapter for MI455X (gfx1250, wave32, WMMA).
//   out = x + up( gelu( down( layernorm(x) ) ) )
// H=1024, bottleneck=64, tokens=8*4096=32768.
// Memory-bound (~256MB HBM traffic): single pass over x, f16 WMMA for the
// two GEMMs (adapter branch only; LN stats / bias / GELU / residual in f32).
// ---------------------------------------------------------------------------

#define H_DIM 1024
#define B_DIM 64
#define TM    64          // tokens per workgroup
#define XS    1032        // padded xn row stride (f16): 516 dwords -> conflict-free b128
#define HS    72          // padded h  row stride (f16): 36 dwords  -> conflict-free b128

typedef __attribute__((ext_vector_type(16))) _Float16 v16h;
typedef __attribute__((ext_vector_type(8)))  _Float16 v8h;
typedef __attribute__((ext_vector_type(8)))  float    v8f;

union V16 { v16h v; v8h h[2]; };

// ---- pre-kernel: fp32 -> fp16 weight conversion into workspace -------------
__global__ void cvt_w_f16(const float* __restrict__ dw,
                          const float* __restrict__ uw,
                          _Float16* __restrict__ dwf,
                          _Float16* __restrict__ uwf, int n) {
  int i = blockIdx.x * blockDim.x + threadIdx.x;
  if (i < n) {
    dwf[i] = (_Float16)dw[i];   // down_w: [64][1024]
    uwf[i] = (_Float16)uw[i];   // up_w:   [1024][64]
  }
}

__global__ void __launch_bounds__(256)
adapter_kernel(const float* __restrict__ x,
               const float* __restrict__ norm_w,
               const float* __restrict__ norm_b,
               const float* __restrict__ down_b,
               const float* __restrict__ up_b,
               const _Float16* __restrict__ dwf,   // [64][1024] f16
               const _Float16* __restrict__ uwf,   // [1024][64] f16
               float* __restrict__ out) {
  extern __shared__ char smem[];
  _Float16* s_xn = (_Float16*)smem;                      // [TM][XS] f16
  _Float16* s_h  = (_Float16*)(smem + TM * XS * 2);      // [TM][HS] f16

  const int tid  = threadIdx.x;
  const int wave = tid >> 5;
  const int lane = tid & 31;
  const int hlf  = lane >> 4;      // which 16-lane half (WMMA layout)
  const int l16  = lane & 15;
  const int tok_base = blockIdx.x * TM;

  // Warm L2 with the (shared, small) f16 weights: 256 threads x 512B each.
  __builtin_prefetch((const char*)dwf + tid * 512);
  __builtin_prefetch((const char*)uwf + tid * 512);

  // ======================= Phase 1: LayerNorm (f32) ========================
  // Each wave handles 8 tokens; lane owns columns [lane*32, lane*32+32).
  float nw[32], nb[32];
  {
    const float4* w4 = (const float4*)(norm_w + lane * 32);
    const float4* b4 = (const float4*)(norm_b + lane * 32);
#pragma unroll
    for (int i = 0; i < 8; ++i) {
      float4 w = w4[i], b = b4[i];
      nw[4*i] = w.x; nw[4*i+1] = w.y; nw[4*i+2] = w.z; nw[4*i+3] = w.w;
      nb[4*i] = b.x; nb[4*i+1] = b.y; nb[4*i+2] = b.z; nb[4*i+3] = b.w;
    }
  }
  for (int t = wave * 8; t < wave * 8 + 8; ++t) {
    const float4* x4 = (const float4*)(x + (size_t)(tok_base + t) * H_DIM + lane * 32);
    float v[32];
#pragma unroll
    for (int i = 0; i < 8; ++i) {
      float4 q = x4[i];
      v[4*i] = q.x; v[4*i+1] = q.y; v[4*i+2] = q.z; v[4*i+3] = q.w;
    }
    float s = 0.f, s2 = 0.f;
#pragma unroll
    for (int i = 0; i < 32; ++i) { s += v[i]; s2 += v[i] * v[i]; }
#pragma unroll
    for (int off = 16; off > 0; off >>= 1) {
      s  += __shfl_xor(s,  off, 32);
      s2 += __shfl_xor(s2, off, 32);
    }
    const float mu   = s * (1.0f / H_DIM);
    const float var  = s2 * (1.0f / H_DIM) - mu * mu;
    const float rstd = rsqrtf(var + 1e-5f);
    __align__(16) _Float16 tmp[32];
#pragma unroll
    for (int i = 0; i < 32; ++i)
      tmp[i] = (_Float16)(((v[i] - mu) * rstd) * nw[i] + nb[i]);
    v8h* dst = (v8h*)(s_xn + t * XS + lane * 32);
#pragma unroll
    for (int i = 0; i < 4; ++i) dst[i] = ((const v8h*)tmp)[i];
  }
  __syncthreads();

  // ================== Phase 2: down-proj (WMMA f16, K=1024) =================
  // wave -> (m-pair p = wave>>2, n-tile nt = wave&3). D tiles: 16x16 each.
  const int p  = wave >> 2;
  const int nt = wave & 3;
  const int m0 = 2 * p, m1 = 2 * p + 1;

  v8f acc0 = {}; v8f acc1 = {};
  // B fragment: lane holds 16 consecutive K of column (nt*16+l16); since
  // down_w is stored [n][k], that column is a contiguous 32B run of row n.
  const _Float16* brow = dwf + (size_t)(nt * 16 + l16) * H_DIM + hlf * 16;
  // A fragment: lane<16 holds K {0..15} of row M=l16; lane>=16 holds +16.
  const _Float16* a0p = s_xn + (m0 * 16 + l16) * XS + hlf * 16;
  const _Float16* a1p = s_xn + (m1 * 16 + l16) * XS + hlf * 16;

  for (int k0 = 0; k0 < H_DIM; k0 += 32) {
    v16h b = *(const v16h*)(brow + k0);
    V16 a0, a1;
    a0.h[0] = *(const v8h*)(a0p + k0);
    a0.h[1] = *(const v8h*)(a0p + k0 + 8);
    a1.h[0] = *(const v8h*)(a1p + k0);
    a1.h[1] = *(const v8h*)(a1p + k0 + 8);
    acc0 = __builtin_amdgcn_wmma_f32_16x16x32_f16(false, a0.v, false, b,
                                                  (short)0, acc0, false, false);
    acc1 = __builtin_amdgcn_wmma_f32_16x16x32_f16(false, a1.v, false, b,
                                                  (short)0, acc1, false, false);
  }
  // bias + exact-erf GELU (f32), store h as f16 to LDS.
  {
    const int col = nt * 16 + l16;
    const float db = down_b[col];
#pragma unroll
    for (int r = 0; r < 8; ++r) {
      float h0 = acc0[r] + db;
      float h1 = acc1[r] + db;
      h0 = 0.5f * h0 * (1.0f + erff(h0 * 0.70710678118654752f));
      h1 = 0.5f * h1 * (1.0f + erff(h1 * 0.70710678118654752f));
      s_h[(m0 * 16 + hlf * 8 + r) * HS + col] = (_Float16)h0;
      s_h[(m1 * 16 + hlf * 8 + r) * HS + col] = (_Float16)h1;
    }
  }
  __syncthreads();

  // =================== Phase 3: up-proj (WMMA f16, K=64) ====================
  // Hoist the (tiny) h A-fragments for both m-tiles and both K-steps.
  V16 ha[2][2];
  {
    const _Float16* h0p = s_h + (m0 * 16 + l16) * HS + hlf * 16;
    const _Float16* h1p = s_h + (m1 * 16 + l16) * HS + hlf * 16;
#pragma unroll
    for (int it = 0; it < 2; ++it) {
      ha[0][it].h[0] = *(const v8h*)(h0p + it * 32);
      ha[0][it].h[1] = *(const v8h*)(h0p + it * 32 + 8);
      ha[1][it].h[0] = *(const v8h*)(h1p + it * 32);
      ha[1][it].h[1] = *(const v8h*)(h1p + it * 32 + 8);
    }
  }
  const int q = wave & 3;
  for (int j = 0; j < 16; ++j) {
    const int n    = q + 4 * j;          // N-tile 0..63
    const int ncol = n * 16 + l16;       // output column for this lane
    const _Float16* br = uwf + (size_t)ncol * B_DIM + hlf * 16;
    v16h b0 = *(const v16h*)(br);        // K  0..31 (lane half-specific)
    v16h b1 = *(const v16h*)(br + 32);   // K 32..63
    v8f c0 = {}; v8f c1 = {};
    c0 = __builtin_amdgcn_wmma_f32_16x16x32_f16(false, ha[0][0].v, false, b0,
                                                (short)0, c0, false, false);
    c0 = __builtin_amdgcn_wmma_f32_16x16x32_f16(false, ha[0][1].v, false, b1,
                                                (short)0, c0, false, false);
    c1 = __builtin_amdgcn_wmma_f32_16x16x32_f16(false, ha[1][0].v, false, b0,
                                                (short)0, c1, false, false);
    c1 = __builtin_amdgcn_wmma_f32_16x16x32_f16(false, ha[1][1].v, false, b1,
                                                (short)0, c1, false, false);
    const float ub = up_b[ncol];
    size_t base0 = (size_t)(tok_base + m0 * 16 + hlf * 8) * H_DIM + ncol;
    size_t base1 = (size_t)(tok_base + m1 * 16 + hlf * 8) * H_DIM + ncol;
#pragma unroll
    for (int r = 0; r < 8; ++r) {
      out[base0 + (size_t)r * H_DIM] = x[base0 + (size_t)r * H_DIM] + c0[r] + ub;
      out[base1 + (size_t)r * H_DIM] = x[base1 + (size_t)r * H_DIM] + c1[r] + ub;
    }
  }
}

extern "C" void kernel_launch(void* const* d_in, const int* in_sizes, int n_in,
                              void* d_out, int out_size, void* d_ws, size_t ws_size,
                              hipStream_t stream) {
  const float* x      = (const float*)d_in[0];
  const float* norm_w = (const float*)d_in[1];
  const float* norm_b = (const float*)d_in[2];
  const float* down_w = (const float*)d_in[3];  // [64][1024]
  const float* down_b = (const float*)d_in[4];
  const float* up_w   = (const float*)d_in[5];  // [1024][64]
  const float* up_b   = (const float*)d_in[6];
  float* out = (float*)d_out;

  _Float16* dwf = (_Float16*)d_ws;              // 64*1024 f16 = 128KB
  _Float16* uwf = dwf + B_DIM * H_DIM;          // 1024*64 f16 = 128KB

  const int nW = B_DIM * H_DIM;                 // 65536 per matrix
  cvt_w_f16<<<(nW + 255) / 256, 256, 0, stream>>>(down_w, up_w, dwf, uwf, nW);

  const int ntok = in_sizes[0] / H_DIM;         // 32768
  const size_t lds_bytes = (size_t)TM * XS * 2 + (size_t)TM * HS * 2;  // 141312
  adapter_kernel<<<ntok / TM, 256, lds_bytes, stream>>>(
      x, norm_w, norm_b, down_b, up_b, dwf, uwf, out);
}
